// GraphEncoder_77283641524506
// MI455X (gfx1250) — compile-verified, matrix operand order checked by reading, not run on hardware
//
#include <hip/hip_runtime.h>

// ---------------- problem constants ----------------
#define NN    20000
#define EE    320000
#define EMB   768
#define HID   256
#define HEADS 8
#define DH    32          // HID / HEADS == wave32 width
#define RR    5
#define BOT   128

typedef __attribute__((ext_vector_type(16))) __bf16 v16bf;
typedef __attribute__((ext_vector_type(8)))  float  v8f;

union Frag16 {
    v16bf v;
    uint4 q[2];
};

// ---------------- helpers ----------------
__device__ __forceinline__ unsigned short f2bf(float f) {
    unsigned u = __float_as_uint(f);
    unsigned r = (u + 0x7FFFu + ((u >> 16) & 1u)) >> 16;   // RNE
    return (unsigned short)r;
}

// monotonic float<->uint mapping for atomic max
__device__ __forceinline__ unsigned ford(float f) {
    unsigned u = __float_as_uint(f);
    return (u & 0x80000000u) ? ~u : (u | 0x80000000u);
}
__device__ __forceinline__ float bford(unsigned u) {
    return (u & 0x80000000u) ? __uint_as_float(u ^ 0x80000000u)
                             : __uint_as_float(~u);
}

// ---------------- elementwise kernels ----------------
__global__ void fill_f32(float* p, float v, size_t n) {
    size_t i = (size_t)blockIdx.x * blockDim.x + threadIdx.x;
    if (i < n) p[i] = v;
}
__global__ void fill_u32(unsigned* p, unsigned v, size_t n) {
    size_t i = (size_t)blockIdx.x * blockDim.x + threadIdx.x;
    if (i < n) p[i] = v;
}
__global__ void cast_bf16(const float* __restrict__ src, unsigned short* __restrict__ dst, size_t n) {
    size_t i = (size_t)blockIdx.x * blockDim.x + threadIdx.x;
    if (i < n) dst[i] = f2bf(src[i]);
}
// Wt[r][n][k] = W[r][k][n]  (cast to bf16)
__global__ void transpose_cast_w(const float* __restrict__ W, unsigned short* __restrict__ Wt,
                                 int Rr, int K, int Nc) {
    size_t i = (size_t)blockIdx.x * blockDim.x + threadIdx.x;
    size_t total = (size_t)Rr * K * Nc;
    if (i >= total) return;
    int n = (int)(i % Nc);
    size_t rk = i / Nc;
    int k = (int)(rk % K);
    int r = (int)(rk / K);
    Wt[((size_t)r * Nc + n) * K + k] = f2bf(W[i]);
}
// h = x + h + row_emb[ri] + col_emb[ci] + mod_emb[mi]
__global__ void combine_embed(const float* __restrict__ x, float* __restrict__ h,
                              const int* __restrict__ ri, const int* __restrict__ ci,
                              const int* __restrict__ mi,
                              const float* __restrict__ rowe, const float* __restrict__ cole,
                              const float* __restrict__ mode) {
    size_t i = (size_t)blockIdx.x * blockDim.x + threadIdx.x;
    if (i >= (size_t)NN * EMB) return;
    int node = (int)(i / EMB);
    int c    = (int)(i % EMB);
    h[i] = x[i] + h[i]
         + rowe[(size_t)ri[node] * EMB + c]
         + cole[(size_t)ci[node] * EMB + c]
         + mode[(size_t)mi[node] * EMB + c];
}
__global__ void elu_inplace(float* x, size_t n) {
    size_t i = (size_t)blockIdx.x * blockDim.x + threadIdx.x;
    if (i < n) {
        float v = x[i];
        x[i] = v > 0.f ? v : (__expf(v) - 1.f);
    }
}

// ---------------- bf16 WMMA GEMM, register-blocked 32x64 per wave ----------------
// C[z][M x Nc] = act(A[M x K] @ B[z]ᵀ + bias), A row-major bf16, Bt = [z][Nc x K] bf16.
// One wave per block; wave owns a 32(M) x 64(N) tile = 2x4 accumulators.
// Per K-step: 12 b128 loads feed 8 v_wmma_f32_16x16x32_bf16.
template <int ACT>   // 0: none, 1: relu
__global__ void gemm_bf16_wmma(const unsigned short* __restrict__ A,
                               const unsigned short* __restrict__ Bt,
                               const float* __restrict__ bias,
                               float* __restrict__ C,
                               int M, int Nc, int K)
{
    const int lane = threadIdx.x & 31;
    const int m0   = blockIdx.x * 32;
    const int n0   = blockIdx.y * 64;
    const int rel  = blockIdx.z;

    const unsigned short* Bz = Bt + (size_t)rel * Nc * K;
    float* Cz = C + (size_t)rel * M * Nc;

    const int l15   = lane & 15;
    const int half8 = (lane >> 4) * 8;                 // K sub-offset per ISA 16-bit layout

    const unsigned short* arow0 = A  + (size_t)(m0 + l15) * K + half8;
    const unsigned short* arow1 = arow0 + (size_t)16 * K;
    const unsigned short* brow0 = Bz + (size_t)(n0 + l15) * K + half8;
    const size_t bstep = (size_t)16 * K;               // next 16-col B tile

    v8f acc[2][4] = {};
    for (int kc = 0; kc < K; kc += 32) {
        __builtin_prefetch(arow0 + kc + 64, 0, 1);
        __builtin_prefetch(arow1 + kc + 64, 0, 1);
        Frag16 a0, a1, b[4];
        a0.q[0] = *reinterpret_cast<const uint4*>(arow0 + kc);
        a0.q[1] = *reinterpret_cast<const uint4*>(arow0 + kc + 16);
        a1.q[0] = *reinterpret_cast<const uint4*>(arow1 + kc);
        a1.q[1] = *reinterpret_cast<const uint4*>(arow1 + kc + 16);
#pragma unroll
        for (int j = 0; j < 4; ++j) {
            const unsigned short* bp = brow0 + (size_t)j * bstep + kc;
            b[j].q[0] = *reinterpret_cast<const uint4*>(bp);
            b[j].q[1] = *reinterpret_cast<const uint4*>(bp + 16);
        }
#pragma unroll
        for (int j = 0; j < 4; ++j) {
            acc[0][j] = __builtin_amdgcn_wmma_f32_16x16x32_bf16(
                            false, a0.v, false, b[j].v, (short)0, acc[0][j], false, false);
            acc[1][j] = __builtin_amdgcn_wmma_f32_16x16x32_bf16(
                            false, a1.v, false, b[j].v, (short)0, acc[1][j], false, false);
        }
    }

    // C layout: VGPR v -> row v (+8 for lanes 16..31), col = lane&15
    const int rsub = (lane >> 4) << 3;
#pragma unroll
    for (int i = 0; i < 2; ++i) {
        const int rbase = m0 + i * 16 + rsub;
#pragma unroll
        for (int j = 0; j < 4; ++j) {
            const int col = n0 + j * 16 + l15;
            const float bv = bias ? bias[col] : 0.f;
#pragma unroll
            for (int v = 0; v < 8; ++v) {
                float val = acc[i][j][v] + bv;
                if (ACT == 1) val = fmaxf(val, 0.f);
                Cz[(size_t)(rbase + v) * Nc + col] = val;
            }
        }
    }
}

// ---------------- edge kernels (one wave per edge; dh == 32 == wave32) ----------------
__global__ void edge_logits(const float* __restrict__ hr,
                            const int* __restrict__ src, const int* __restrict__ dst,
                            const int* __restrict__ etype,
                            const float* __restrict__ a_src, const float* __restrict__ a_dst,
                            float* __restrict__ logit, unsigned* __restrict__ mbits)
{
    const int lane = threadIdx.x & 31;
    const int e = blockIdx.x * (blockDim.x >> 5) + (threadIdx.x >> 5);
    if (e >= EE) return;
    const int et = etype[e], s = src[e], d = dst[e];
    const float* zs = hr + ((size_t)et * NN + s) * HID;
    const float* zd = hr + ((size_t)et * NN + d) * HID;
#pragma unroll
    for (int h = 0; h < HEADS; ++h) {
        const int idx = h * DH + lane;
        float v = zs[idx] * a_src[(et * HEADS + h) * DH + lane]
                + zd[idx] * a_dst[(et * HEADS + h) * DH + lane];
#pragma unroll
        for (int off = 16; off; off >>= 1) v += __shfl_xor(v, off, 32);
        if (lane == 0) {
            float l = v > 0.f ? v : 0.2f * v;          // leaky_relu, slope 0.2
            logit[(size_t)e * HEADS + h] = l;
            atomicMax(&mbits[(size_t)d * HEADS + h], ford(l));
        }
    }
}

__global__ void exp_and_sum(const float* __restrict__ logit, const int* __restrict__ dst,
                            const unsigned* __restrict__ mbits,
                            float* __restrict__ att, float* __restrict__ ssum)
{
    size_t i = (size_t)blockIdx.x * blockDim.x + threadIdx.x;
    if (i >= (size_t)EE * HEADS) return;
    int e = (int)(i >> 3), h = (int)(i & 7);
    int d = dst[e];
    float m  = bford(mbits[(size_t)d * HEADS + h]);
    float ev = __expf(logit[i] - m);
    att[i] = ev;
    atomicAdd(&ssum[(size_t)d * HEADS + h], ev);
}

__global__ void aggregate(const float* __restrict__ hr,
                          const float* __restrict__ att, const float* __restrict__ ssum,
                          const int* __restrict__ src, const int* __restrict__ dst,
                          const int* __restrict__ etype,
                          float* __restrict__ out)
{
    const int lane = threadIdx.x & 31;
    const int e = blockIdx.x * (blockDim.x >> 5) + (threadIdx.x >> 5);
    if (e >= EE) return;
    const int et = etype[e], s = src[e], d = dst[e];
    const float* zs = hr + ((size_t)et * NN + s) * HID;
#pragma unroll
    for (int h = 0; h < HEADS; ++h) {
        float alpha = att[(size_t)e * HEADS + h]
                    / (ssum[(size_t)d * HEADS + h] + 1e-16f);
        atomicAdd(&out[(size_t)d * HID + h * DH + lane], alpha * zs[h * DH + lane]);
    }
}

// ---------------- host orchestration ----------------
static inline dim3 grid1d(size_t n, int b) { return dim3((unsigned)((n + b - 1) / b)); }

extern "C" void kernel_launch(void* const* d_in, const int* in_sizes, int n_in,
                              void* d_out, int out_size, void* d_ws, size_t ws_size,
                              hipStream_t stream)
{
    const float* x      = (const float*)d_in[0];
    const int*   ri     = (const int*)  d_in[1];
    const int*   ci     = (const int*)  d_in[2];
    const int*   mi     = (const int*)  d_in[3];
    const int*   eidx   = (const int*)  d_in[4];
    const int*   etype  = (const int*)  d_in[5];
    const float* w_down = (const float*)d_in[6];
    const float* b_down = (const float*)d_in[7];
    const float* w_up   = (const float*)d_in[8];
    const float* b_up   = (const float*)d_in[9];
    const float* rowe   = (const float*)d_in[10];
    const float* cole   = (const float*)d_in[11];
    const float* mode   = (const float*)d_in[12];
    const float* w0     = (const float*)d_in[13];
    const float* as0    = (const float*)d_in[14];
    const float* ad0    = (const float*)d_in[15];
    const float* w1     = (const float*)d_in[16];
    const float* as1    = (const float*)d_in[17];
    const float* ad1    = (const float*)d_in[18];
    float* out = (float*)d_out;
    const int* src = eidx;
    const int* dst = eidx + EE;

    // ---- workspace carve-up (floats) ----
    float* wsf   = (float*)d_ws;
    float* hbuf  = wsf;                               // N*EMB
    float* tbuf  = hbuf + (size_t)NN * EMB;           // N*BOT
    float* hr    = tbuf + (size_t)NN * BOT;           // R*N*HID
    float* att   = hr   + (size_t)RR * NN * HID;      // E*HEADS
    unsigned* mbits = (unsigned*)(att + (size_t)EE * HEADS);   // N*HEADS
    float* ssum  = (float*)(mbits + (size_t)NN * HEADS);       // N*HEADS
    float* hout  = ssum + (size_t)NN * HEADS;         // N*HID
    unsigned short* abf = (unsigned short*)(hout + (size_t)NN * HID); // N*EMB bf16
    unsigned short* wbf = abf + (size_t)NN * EMB;     // up to R*EMB*HID bf16

    const int TB = 256;
    const dim3 gemmBlk(32);                           // one wave per block
    const unsigned gemmGx = NN / 32;                  // 625 M-blocks (exact)

    // ===== Adapter: t = relu(x @ w_down + b_down) =====
    cast_bf16<<<grid1d((size_t)NN * EMB, TB), TB, 0, stream>>>(x, abf, (size_t)NN * EMB);
    transpose_cast_w<<<grid1d((size_t)EMB * BOT, TB), TB, 0, stream>>>(w_down, wbf, 1, EMB, BOT);
    gemm_bf16_wmma<1><<<dim3(gemmGx, BOT / 64, 1), gemmBlk, 0, stream>>>(abf, wbf, b_down, tbuf, NN, BOT, EMB);

    // ===== Adapter: u = t @ w_up + b_up  -> hbuf =====
    cast_bf16<<<grid1d((size_t)NN * BOT, TB), TB, 0, stream>>>(tbuf, abf, (size_t)NN * BOT);
    transpose_cast_w<<<grid1d((size_t)BOT * EMB, TB), TB, 0, stream>>>(w_up, wbf, 1, BOT, EMB);
    gemm_bf16_wmma<0><<<dim3(gemmGx, EMB / 64, 1), gemmBlk, 0, stream>>>(abf, wbf, b_up, hbuf, NN, EMB, BOT);

    // ===== h = x + u + embeddings =====
    combine_embed<<<grid1d((size_t)NN * EMB, TB), TB, 0, stream>>>(x, hbuf, ri, ci, mi, rowe, cole, mode);

    // ===== RGAT layer 0: hr[r] = h @ W0[r] =====
    cast_bf16<<<grid1d((size_t)NN * EMB, TB), TB, 0, stream>>>(hbuf, abf, (size_t)NN * EMB);
    transpose_cast_w<<<grid1d((size_t)RR * EMB * HID, TB), TB, 0, stream>>>(w0, wbf, RR, EMB, HID);
    gemm_bf16_wmma<0><<<dim3(gemmGx, HID / 64, RR), gemmBlk, 0, stream>>>(abf, wbf, nullptr, hr, NN, HID, EMB);

    // ---- edge phase 0 ----
    fill_u32<<<grid1d((size_t)NN * HEADS, TB), TB, 0, stream>>>(mbits, 0u, (size_t)NN * HEADS);
    fill_f32<<<grid1d((size_t)NN * HEADS, TB), TB, 0, stream>>>(ssum, 0.f, (size_t)NN * HEADS);
    fill_f32<<<grid1d((size_t)NN * HID, TB), TB, 0, stream>>>(hout, 0.f, (size_t)NN * HID);
    edge_logits<<<dim3(EE / 8), dim3(256), 0, stream>>>(hr, src, dst, etype, as0, ad0, att, mbits);
    exp_and_sum<<<grid1d((size_t)EE * HEADS, TB), TB, 0, stream>>>(att, dst, mbits, att, ssum);
    aggregate<<<dim3(EE / 8), dim3(256), 0, stream>>>(hr, att, ssum, src, dst, etype, hout);
    elu_inplace<<<grid1d((size_t)NN * HID, TB), TB, 0, stream>>>(hout, (size_t)NN * HID);

    // ===== RGAT layer 1: hr[r] = h1 @ W1[r] =====
    cast_bf16<<<grid1d((size_t)NN * HID, TB), TB, 0, stream>>>(hout, abf, (size_t)NN * HID);
    transpose_cast_w<<<grid1d((size_t)RR * HID * HID, TB), TB, 0, stream>>>(w1, wbf, RR, HID, HID);
    gemm_bf16_wmma<0><<<dim3(gemmGx, HID / 64, RR), gemmBlk, 0, stream>>>(abf, wbf, nullptr, hr, NN, HID, HID);

    // ---- edge phase 1 (writes final output) ----
    fill_u32<<<grid1d((size_t)NN * HEADS, TB), TB, 0, stream>>>(mbits, 0u, (size_t)NN * HEADS);
    fill_f32<<<grid1d((size_t)NN * HEADS, TB), TB, 0, stream>>>(ssum, 0.f, (size_t)NN * HEADS);
    fill_f32<<<grid1d((size_t)NN * HID, TB), TB, 0, stream>>>(out, 0.f, (size_t)NN * HID);
    edge_logits<<<dim3(EE / 8), dim3(256), 0, stream>>>(hr, src, dst, etype, as1, ad1, att, mbits);
    exp_and_sum<<<grid1d((size_t)EE * HEADS, TB), TB, 0, stream>>>(att, dst, mbits, att, ssum);
    aggregate<<<dim3(EE / 8), dim3(256), 0, stream>>>(hr, att, ssum, src, dst, etype, out);
}
